// transformer_49168785604995
// MI455X (gfx1250) — matrix operand order repaired and checked by reading, not compiled, not run on hardware
//
#include <hip/hip_runtime.h>

#define B_     8
#define N_     1516
#define CIN_   128
#define COUT_  512
#define K_     20
#define HEADS_ 8
#define DIMK_  32
#define CH_    128
#define DIMV_  64
#define CC_    320            // 2*CIN + DIMV

#define XYZ_ELEMS (B_*N_*3)   // 36384

typedef __attribute__((ext_vector_type(16))) _Float16 v16h;
typedef __attribute__((ext_vector_type(8)))  _Float16 v8h;
typedef __attribute__((ext_vector_type(8)))  float    v8f;

// ---------------- workspace layout (bytes) ----------------
#define OFF_IDX  0u                         // int[B*N*K]            = 970240 B
#define OFF_W1H  970240u                    // f16[128*320]          = 81920 B
#define OFF_W2H  1052160u                   // f16[128*128]          = 32768 B
#define OFF_WVH  1084928u                   // f16[64*320]           = 40960 B
#define OFF_WKH  1125888u                   // f16[32*320]           = 20480 B
#define OFF_BN1  1146368u                   // f32[256]  (scale|shift)
#define OFF_BN2  1147392u                   // f32[256]
#define OFF_BNV  1148416u                   // f32[128]
#define OFF_BNQ  1148928u                   // f32[512]

// row strides (in f16 elements) for LDS tiles; both *2 bytes are multiples of 16
#define CAT_RS 328
#define H1_RS  136

// -------- A-fragment loader: 16-bit A 16x32 per ISA layout --------
// p must point at (row m of this lane) + kbase + hi*8   (hi = lane>>4)
__device__ __forceinline__ v16h loadA(const _Float16* p) {
  v8h lo  = *(const v8h*)p;          // K = kbase + hi*8 + 0..7
  v8h hi8 = *(const v8h*)(p + 16);   // K = kbase + 16 + hi*8 + 0..7
  v16h a;
#pragma unroll
  for (int i = 0; i < 8; ++i) { a[i] = lo[i]; a[i + 8] = hi8[i]; }
  return a;
}

// ---------------- prep: f16 weights, bn scale/shift, xyz copy ----------------
__global__ void prep_kernel(const float* __restrict__ xyz,
                            const float* __restrict__ W1, const float* __restrict__ W2,
                            const float* __restrict__ Wv, const float* __restrict__ Wk,
                            const float* __restrict__ g1, const float* __restrict__ b1,
                            const float* __restrict__ m1, const float* __restrict__ v1,
                            const float* __restrict__ g2, const float* __restrict__ b2,
                            const float* __restrict__ m2, const float* __restrict__ v2,
                            const float* __restrict__ gv, const float* __restrict__ bv,
                            const float* __restrict__ mv, const float* __restrict__ vv,
                            const float* __restrict__ gq, const float* __restrict__ bq,
                            const float* __restrict__ mq, const float* __restrict__ vq,
                            _Float16* __restrict__ W1h, _Float16* __restrict__ W2h,
                            _Float16* __restrict__ Wvh, _Float16* __restrict__ Wkh,
                            float* __restrict__ bn1p, float* __restrict__ bn2p,
                            float* __restrict__ bnvp, float* __restrict__ bnqp,
                            float* __restrict__ out)
{
  const int total = 40960 + 16384 + 20480 + 10240 + 128 + 128 + 64 + 256 + XYZ_ELEMS;
  for (int i = blockIdx.x * blockDim.x + threadIdx.x; i < total; i += gridDim.x * blockDim.x) {
    int j = i;
    if (j < 40960) { W1h[j] = (_Float16)W1[j]; continue; } j -= 40960;
    if (j < 16384) { W2h[j] = (_Float16)W2[j]; continue; } j -= 16384;
    if (j < 20480) { Wvh[j] = (_Float16)Wv[j]; continue; } j -= 20480;
    if (j < 10240) { Wkh[j] = (_Float16)Wk[j]; continue; } j -= 10240;
    if (j < 128) { float s = g1[j]*rsqrtf(v1[j]+1e-5f); bn1p[j]=s; bn1p[128+j]=b1[j]-m1[j]*s; continue; } j -= 128;
    if (j < 128) { float s = g2[j]*rsqrtf(v2[j]+1e-5f); bn2p[j]=s; bn2p[128+j]=b2[j]-m2[j]*s; continue; } j -= 128;
    if (j < 64)  { float s = gv[j]*rsqrtf(vv[j]+1e-5f); bnvp[j]=s; bnvp[64+j]=bv[j]-mv[j]*s; continue; } j -= 64;
    if (j < 256) { float s = gq[j]*rsqrtf(vq[j]+1e-5f); bnqp[j]=s; bnqp[256+j]=bq[j]-mq[j]*s; continue; } j -= 256;
    out[j] = xyz[j];  // xyz pass-through (first tuple element)
  }
}

// ---------------- KNN: per point, register-resident top-20 ----------------
__global__ __launch_bounds__(256) void knn_kernel(const float* __restrict__ xyz,
                                                  int* __restrict__ idxg)
{
  __shared__ float sx[256 * 3];
  const int tid = threadIdx.x;
  const int n = blockIdx.x * 256 + tid;
  const int b = blockIdx.y;
  const size_t bN = (size_t)b * N_;
  const bool valid = (n < N_);
  float px = 0.f, py = 0.f, pz = 0.f;
  if (valid) { px = xyz[(bN+n)*3]; py = xyz[(bN+n)*3+1]; pz = xyz[(bN+n)*3+2]; }

  float bd[K_]; int bi[K_];
#pragma unroll
  for (int q = 0; q < K_; ++q) { bd[q] = 3.4e38f; bi[q] = 0; }

  for (int m0 = 0; m0 < N_; m0 += 256) {
    const int mm = m0 + tid;
    if (mm < N_) {
      sx[tid*3]   = xyz[(bN+mm)*3];
      sx[tid*3+1] = xyz[(bN+mm)*3+1];
      sx[tid*3+2] = xyz[(bN+mm)*3+2];
    }
    __syncthreads();
    const int lim = (N_ - m0) < 256 ? (N_ - m0) : 256;
    for (int j = 0; j < lim; ++j) {
      const float dx = px - sx[j*3], dy = py - sx[j*3+1], dz = pz - sx[j*3+2];
      const float d = dx*dx + dy*dy + dz*dz;   // same ordering as ||x||^2+||y||^2-2x.y
      if (valid && d < bd[K_-1]) {
        float cd = d; int ci = m0 + j;
#pragma unroll
        for (int q = 0; q < K_; ++q) {
          if (cd < bd[q]) { float tf = bd[q]; bd[q] = cd; cd = tf;
                            int   ti = bi[q]; bi[q] = ci; ci = ti; }
        }
      }
    }
    __syncthreads();
  }
  if (valid) {
#pragma unroll
    for (int q = 0; q < K_; ++q) idxg[(bN+n)*K_ + q] = bi[q];
  }
}

// ---------------- fused per-point pipeline (one workgroup per (b,n)) ----------------
__global__ __launch_bounds__(256) void fused_kernel(
    const float* __restrict__ xyz, const float* __restrict__ feat,
    const float* __restrict__ Wh,  const float* __restrict__ Wq,
    const int* __restrict__ idxg,
    const _Float16* __restrict__ W1h, const _Float16* __restrict__ W2h,
    const _Float16* __restrict__ Wvh, const _Float16* __restrict__ Wkh,
    const float* __restrict__ bn1p, const float* __restrict__ bn2p,
    const float* __restrict__ bnvp, const float* __restrict__ bnqp,
    float* __restrict__ out)
{
  __shared__ __align__(16) _Float16 s_cat[32 * CAT_RS];   // 20.5 KB, rows 20..31 zero
  __shared__ __align__(16) _Float16 s_h1[32 * H1_RS];     // 8.5 KB
  __shared__ float s_fq[CH_];
  __shared__ float s_q[256];
  __shared__ float s_att[32 * 32];
  __shared__ float s_val[32 * 64];
  __shared__ float s_kv[DIMK_ * DIMV_];
  __shared__ int   s_idx[K_];
  __shared__ float s_rel[K_ * 3];
  __shared__ float s_cen[CIN_];
  __shared__ float s_wh[DIMV_ * 3];

  const int tid  = threadIdx.x;
  const int lane = tid & 31;
  const int wave = tid >> 5;
  const int n = blockIdx.x;
  const int b = blockIdx.y;
  const size_t bN = (size_t)b * N_;
  const int mlane = lane & 15;
  const int hi    = lane >> 4;

  // ---- stage 0: gather indices / center / rel / Wh ----
  if (tid < K_) s_idx[tid] = idxg[(bN + n)*K_ + tid];
  for (int e = tid; e < DIMV_*3; e += 256) s_wh[e] = Wh[e];
  for (int e = tid; e < CIN_;    e += 256) s_cen[e] = feat[(bN+n)*CIN_ + e];
  __syncthreads();
  if (tid < K_*3) {
    int k = tid / 3, d = tid - k*3;
    s_rel[tid] = xyz[(bN + s_idx[k])*3 + d] - xyz[(bN + n)*3 + d];
  }
  __syncthreads();

  // ---- stage 1: build cat[32 x 320] (f16) ----
  for (int e = tid; e < 32 * CC_; e += 256) {
    int k = e / CC_, c = e - k * CC_;
    float v = 0.f;
    if (k < K_) {
      if (c < CIN_)            v = feat[(bN + s_idx[k])*CIN_ + c] - s_cen[c];
      else if (c < 2*CIN_)     v = s_cen[c - CIN_];
      else { int vv = c - 2*CIN_;
             v = s_wh[vv*3]*s_rel[k*3] + s_wh[vv*3+1]*s_rel[k*3+1] + s_wh[vv*3+2]*s_rel[k*3+2]; }
    }
    s_cat[k*CAT_RS + c] = (_Float16)v;
  }
  __syncthreads();

  // ---- stage 2: h1 = relu(bn1(W1 @ cat))  [32 x 128], WMMA ----
  {
    const int nt = wave;                       // 8 N-tiles of 16
    v8f acc0 = {}; v8f acc1 = {};
    for (int kb = 0; kb < CC_; kb += 32) {
      v16h bfrag = loadA(W1h + (nt*16 + mlane)*CC_ + kb + hi*8);
      v16h a0 = loadA(s_cat + mlane*CAT_RS + kb + hi*8);
      v16h a1 = loadA(s_cat + (16 + mlane)*CAT_RS + kb + hi*8);
      acc0 = __builtin_amdgcn_wmma_f32_16x16x32_f16(false, a0, false, bfrag, (short)0, acc0, false, false);
      acc1 = __builtin_amdgcn_wmma_f32_16x16x32_f16(false, a1, false, bfrag, (short)0, acc1, false, false);
    }
    const int o = nt*16 + mlane;
    const float s = bn1p[o], t = bn1p[128 + o];
#pragma unroll
    for (int r = 0; r < 8; ++r) {
      const int m0 = r + hi*8;
      s_h1[m0*H1_RS + o]        = (_Float16)fmaxf(0.f, acc0[r]*s + t);
      s_h1[(16 + m0)*H1_RS + o] = (_Float16)fmaxf(0.f, acc1[r]*s + t);
    }
  }
  __syncthreads();

  // ---- stage 3: h2 = relu(bn2(W2 @ h1)); fq = max_k h2  (WMMA + masked max) ----
  {
    const int nt = wave;
    v8f acc0 = {}; v8f acc1 = {};
    for (int kb = 0; kb < CH_; kb += 32) {
      v16h bfrag = loadA(W2h + (nt*16 + mlane)*CH_ + kb + hi*8);
      v16h a0 = loadA(s_h1 + mlane*H1_RS + kb + hi*8);
      v16h a1 = loadA(s_h1 + (16 + mlane)*H1_RS + kb + hi*8);
      acc0 = __builtin_amdgcn_wmma_f32_16x16x32_f16(false, a0, false, bfrag, (short)0, acc0, false, false);
      acc1 = __builtin_amdgcn_wmma_f32_16x16x32_f16(false, a1, false, bfrag, (short)0, acc1, false, false);
    }
    const int o = nt*16 + mlane;
    const float s = bn2p[o], t = bn2p[128 + o];
    float pm = 0.f;                                  // relu >= 0
#pragma unroll
    for (int r = 0; r < 8; ++r) {
      const int m0 = r + hi*8;                       // < 16, always a real row
      pm = fmaxf(pm, fmaxf(0.f, acc0[r]*s + t));
      const int m1 = 16 + m0;
      const float x1 = fmaxf(0.f, acc1[r]*s + t);
      if (m1 < K_) pm = fmaxf(pm, x1);               // mask padded rows
    }
    pm = fmaxf(pm, __shfl_xor(pm, 16, 32));          // combine the two half-wave row groups
    if (lane < 16) s_fq[o] = pm;                     // each channel owned by exactly one lane
  }
  __syncthreads();

  // ---- stage 4a: q = relu(bnq(Wq @ fq))  (VALU, 256 channels) ----
  {
    float accq = 0.f;
    const float* wr = Wq + tid * CH_;
    for (int c = 0; c < CH_; ++c) accq += wr[c] * s_fq[c];
    s_q[tid] = fmaxf(0.f, accq * bnqp[tid] + bnqp[256 + tid]);
  }
  // ---- stage 4b: val = bnv(Wv @ cat)  [32 x 64], WMMA ----
  {
    const int mt = wave & 1, vt = wave >> 1;         // 2 x 4 tiles over 8 waves
    v8f acc = {};
    for (int kb = 0; kb < CC_; kb += 32) {
      v16h bfrag = loadA(Wvh + (vt*16 + mlane)*CC_ + kb + hi*8);
      v16h a = loadA(s_cat + (mt*16 + mlane)*CAT_RS + kb + hi*8);
      acc = __builtin_amdgcn_wmma_f32_16x16x32_f16(false, a, false, bfrag, (short)0, acc, false, false);
    }
    const int o = vt*16 + mlane;
    const float s = bnvp[o], t = bnvp[64 + o];
#pragma unroll
    for (int r = 0; r < 8; ++r) {
      const int m = mt*16 + r + hi*8;
      s_val[m*64 + o] = acc[r]*s + t;
    }
  }
  // ---- stage 4c: logits = Wk @ cat  [32 x 32], WMMA on waves 0..3 ----
  if (wave < 4) {
    const int mt = wave & 1, dt = wave >> 1;         // 2 x 2 tiles
    v8f acc = {};
    for (int kb = 0; kb < CC_; kb += 32) {
      v16h bfrag = loadA(Wkh + (dt*16 + mlane)*CC_ + kb + hi*8);
      v16h a = loadA(s_cat + (mt*16 + mlane)*CAT_RS + kb + hi*8);
      acc = __builtin_amdgcn_wmma_f32_16x16x32_f16(false, a, false, bfrag, (short)0, acc, false, false);
    }
#pragma unroll
    for (int r = 0; r < 8; ++r) {
      const int m = mt*16 + r + hi*8;
      s_att[m*32 + dt*16 + mlane] = acc[r];
    }
  }
  __syncthreads();

  // ---- stage 5: softmax over k (20 real rows) per d ----
  if (tid < DIMK_) {
    float mx = -3.4e38f;
    for (int k = 0; k < K_; ++k) mx = fmaxf(mx, s_att[k*32 + tid]);
    float sum = 0.f;
    for (int k = 0; k < K_; ++k) sum += __expf(s_att[k*32 + tid] - mx);
    const float inv = 1.f / sum;
    for (int k = 0; k < K_; ++k) s_att[k*32 + tid] = __expf(s_att[k*32 + tid] - mx) * inv;
  }
  __syncthreads();

  // ---- stage 6: kv[d][v] = sum_k kw[k][d] * val[k][v]  (VALU) ----
  for (int e = tid; e < DIMK_*DIMV_; e += 256) {
    const int d = e >> 6, v = e & 63;
    float a = 0.f;
    for (int k = 0; k < K_; ++k) a += s_att[k*32 + d] * s_val[k*64 + v];
    s_kv[e] = a;
  }
  __syncthreads();

  // ---- stage 7: out[h][v] = sum_d q[h*32+d] * kv[d][v]; write transposed (b, c, n) ----
  for (int e = tid; e < COUT_; e += 256) {
    const int h = e >> 6, v = e & 63;
    float a = 0.f;
    for (int d = 0; d < DIMK_; ++d) a += s_q[h*DIMK_ + d] * s_kv[d*DIMV_ + v];
    out[XYZ_ELEMS + ((size_t)b*COUT_ + e)*N_ + n] = a;
  }
}

extern "C" void kernel_launch(void* const* d_in, const int* in_sizes, int n_in,
                              void* d_out, int out_size, void* d_ws, size_t ws_size,
                              hipStream_t stream) {
  (void)in_sizes; (void)n_in; (void)out_size; (void)ws_size;
  const float* xyz  = (const float*)d_in[0];
  const float* feat = (const float*)d_in[1];
  const float* Wh   = (const float*)d_in[2];
  const float* W1   = (const float*)d_in[3];
  const float* W2   = (const float*)d_in[4];
  const float* Wv   = (const float*)d_in[5];
  const float* Wk   = (const float*)d_in[6];
  const float* Wq   = (const float*)d_in[7];
  const float* g1 = (const float*)d_in[8],  *b1 = (const float*)d_in[9];
  const float* m1 = (const float*)d_in[10], *v1 = (const float*)d_in[11];
  const float* g2 = (const float*)d_in[12], *b2 = (const float*)d_in[13];
  const float* m2 = (const float*)d_in[14], *v2 = (const float*)d_in[15];
  const float* gv = (const float*)d_in[16], *bv = (const float*)d_in[17];
  const float* mv = (const float*)d_in[18], *vv = (const float*)d_in[19];
  const float* gq = (const float*)d_in[20], *bq = (const float*)d_in[21];
  const float* mq = (const float*)d_in[22], *vq = (const float*)d_in[23];

  float* out = (float*)d_out;
  char* ws = (char*)d_ws;
  int*      idx  = (int*)(ws + OFF_IDX);
  _Float16* W1h  = (_Float16*)(ws + OFF_W1H);
  _Float16* W2h  = (_Float16*)(ws + OFF_W2H);
  _Float16* Wvh  = (_Float16*)(ws + OFF_WVH);
  _Float16* Wkh  = (_Float16*)(ws + OFF_WKH);
  float*    bn1p = (float*)(ws + OFF_BN1);
  float*    bn2p = (float*)(ws + OFF_BN2);
  float*    bnvp = (float*)(ws + OFF_BNV);
  float*    bnqp = (float*)(ws + OFF_BNQ);

  prep_kernel<<<128, 256, 0, stream>>>(xyz, W1, W2, Wv, Wk,
                                       g1,b1,m1,v1, g2,b2,m2,v2,
                                       gv,bv,mv,vv, gq,bq,mq,vq,
                                       W1h, W2h, Wvh, Wkh,
                                       bn1p, bn2p, bnvp, bnqp, out);
  knn_kernel<<<dim3((N_ + 255)/256, B_), 256, 0, stream>>>(xyz, idx);
  fused_kernel<<<dim3(N_, B_), 256, 0, stream>>>(xyz, feat, Wh, Wq, idx,
                                                 W1h, W2h, Wvh, Wkh,
                                                 bn1p, bn2p, bnvp, bnqp, out);
}
